// GCN_Module_86827058855971
// MI455X (gfx1250) — matrix operand order repaired and checked by reading, not compiled, not run on hardware
//
#include <hip/hip_runtime.h>
#include <hip/hip_bf16.h>
#include <cstdint>

// ---------------------------------------------------------------------------
// GCN module for MI455X (gfx1250), wave32 + WMMA f32_16x16x32_f16 +
// async global->LDS staging (ASYNCcnt) for the P·X B-operands.
// ---------------------------------------------------------------------------

typedef __attribute__((ext_vector_type(16))) _Float16 v16h;
typedef __attribute__((ext_vector_type(8)))  _Float16 v8h;
typedef __attribute__((ext_vector_type(8)))  float    v8f;

static constexpr int NN   = 4096;   // boxes
static constexpr int NFG  = 256;    // feature dim
static constexpr int NFR  = 64;     // relation dim
static constexpr int NG   = 4;      // graphs
static constexpr float RADIUS2 = 0.16f;     // (0.005*80)^2
static constexpr float SCALE   = 0.125f;    // 1/sqrt(64)
static constexpr float LN_EPS  = 1e-5f;

// ---- WMMA helpers ----------------------------------------------------------

__device__ __forceinline__ v8f wmma_f16(v16h a, v16h b, v8f c) {
  // D = A(16x32 f16) * B(32x16 f16) + C(16x16 f32)
  return __builtin_amdgcn_wmma_f32_16x16x32_f16(false, a, false, b,
                                                (short)0, c, false, false);
}

// A operand: 16x32 tile from row-major f16 matrix (row stride ld halves).
// Lane layout: lane lo=row; halves i -> K = k0 + 16*hi + i
// (matches load_b16).  => two contiguous 16B chunks.
__device__ __forceinline__ v16h load_a16(const _Float16* base, int ld,
                                         int row0, int k0, int lane) {
  int lo = lane & 15, hi = lane >> 4;
  const _Float16* p = base + (size_t)(row0 + lo) * ld + k0 + 16 * hi;
  v8h x = *(const v8h*)(p);
  v8h y = *(const v8h*)(p + 8);
  return __builtin_shufflevector(x, y, 0,1,2,3,4,5,6,7,8,9,10,11,12,13,14,15);
}

// B operand: 32x16 tile where element (k,n) = BT[n][k], BT row-major with
// stride ld halves.  Lane lo = column n; halves i -> K = k0 + 16*hi + i.
// => one contiguous 32B chunk per lane.
__device__ __forceinline__ v16h load_b16(const _Float16* baseT, int ld,
                                         int col0, int k0, int lane) {
  int lo = lane & 15, hi = lane >> 4;
  const _Float16* p = baseT + (size_t)(col0 + lo) * ld + k0 + 16 * hi;
  return *(const v16h*)(p);
}

// ---- Prep kernels ----------------------------------------------------------

__global__ void pos_kernel(const float* __restrict__ boxes,
                           float* __restrict__ pos) {
  int n = blockIdx.x * blockDim.x + threadIdx.x;
  if (n < NN) {
    pos[2 * n]     = (boxes[4 * n + 0] + boxes[4 * n + 2]) * 0.5f;
    pos[2 * n + 1] = (boxes[4 * n + 1] + boxes[4 * n + 3]) * 0.5f;
  }
}

__global__ void cvt_kernel(const float* __restrict__ in,
                           _Float16* __restrict__ out, int n) {
  int i = blockIdx.x * blockDim.x + threadIdx.x;
  if (i < n) out[i] = (_Float16)in[i];
}

// X -> f16 row-major and f16 transposed [NFG][N]
__global__ void cvtXT_kernel(const float* __restrict__ X,
                             _Float16* __restrict__ Xh,
                             _Float16* __restrict__ XT) {
  int i = blockIdx.x * blockDim.x + threadIdx.x;  // N*NFG threads
  int n = i >> 8, f = i & 255;
  float v = X[i];
  Xh[i] = (_Float16)v;
  XT[(size_t)f * NN + n] = (_Float16)v;
}

// ---- th/ph projection: out[g][n][r] = X[n]·W[g][r] + b[g][r]  -------------

__global__ __launch_bounds__(128) void thph_kernel(
    const _Float16* __restrict__ Xh, const _Float16* __restrict__ Wh,
    const float* __restrict__ bias, _Float16* __restrict__ outh) {
  int wave = threadIdx.x >> 5, lane = threadIdx.x & 31;
  int lo = lane & 15, hi = lane >> 4;
  int g  = blockIdx.y;
  int n0 = blockIdx.x * 64 + wave * 16;
  const _Float16* W = Wh + (size_t)g * NFR * NFG;   // [64][256] row-major

  v8f acc[4] = {};
  for (int kk = 0; kk < NFG; kk += 32) {
    v16h a = load_a16(Xh, NFG, n0, kk, lane);
#pragma unroll
    for (int rt = 0; rt < 4; ++rt) {
      v16h b = load_b16(W, NFG, rt * 16, kk, lane);  // B(k,r)=W[r][k]
      acc[rt] = wmma_f16(a, b, acc[rt]);
    }
  }
#pragma unroll
  for (int rt = 0; rt < 4; ++rt) {
    int r = rt * 16 + lo;
    float bv = bias[g * NFR + r];
#pragma unroll
    for (int j = 0; j < 8; ++j) {
      int n = n0 + j + 8 * hi;
      outh[((size_t)g * NN + n) * NFR + r] = (_Float16)(acc[rt][j] + bv);
    }
  }
}

// ---- softmax denominator: rowinv[g][n] = 1 / sum_m exp(s_nm) (masked=0) ---

__global__ __launch_bounds__(128) void denom_kernel(
    const _Float16* __restrict__ th, const _Float16* __restrict__ ph,
    const float* __restrict__ pos, float* __restrict__ rowinv) {
  int wave = threadIdx.x >> 5, lane = threadIdx.x & 31;
  int lo = lane & 15, hi = lane >> 4;
  int g  = blockIdx.y;
  int n0 = blockIdx.x * 64 + wave * 16;
  const _Float16* thg = th + (size_t)g * NN * NFR;
  const _Float16* phg = ph + (size_t)g * NN * NFR;

  v16h a0 = load_a16(thg, NFR, n0, 0,  lane);
  v16h a1 = load_a16(thg, NFR, n0, 32, lane);

  float px[8], py[8];
#pragma unroll
  for (int j = 0; j < 8; ++j) {
    int n = n0 + j + 8 * hi;
    px[j] = pos[2 * n]; py[j] = pos[2 * n + 1];
  }

  v8f sacc = {};
  for (int m0 = 0; m0 < NN; m0 += 16) {
    v16h b0 = load_b16(phg, NFR, m0, 0,  lane);   // B(r,m)=ph[m][r]
    v16h b1 = load_b16(phg, NFR, m0, 32, lane);
    v8f c = {};
    c = wmma_f16(a0, b0, c);
    c = wmma_f16(a1, b1, c);
    float qx = pos[2 * (m0 + lo)], qy = pos[2 * (m0 + lo) + 1];
#pragma unroll
    for (int j = 0; j < 8; ++j) {
      float dx = px[j] - qx, dy = py[j] - qy;
      float e = (dx * dx + dy * dy > RADIUS2) ? 0.0f : __expf(c[j] * SCALE);
      sacc[j] += e;
    }
  }
  // reduce across the 16 lanes holding one row's columns
#pragma unroll
  for (int j = 0; j < 8; ++j) {
    float v = sacc[j];
    for (int off = 1; off < 16; off <<= 1) v += __shfl_xor(v, off, 32);
    if (lo == 0) rowinv[g * NN + n0 + j + 8 * hi] = 1.0f / v;
  }
}

// ---- async stage of one XT chunk [256 features][16 of 32 m] into LDS ------------
// 8 KB per chunk, 128 threads, 4 x b128 per thread, tracked by ASYNCcnt.

__device__ __forceinline__ void fill_xbuf_async(_Float16* dst_lds,
                                                const _Float16* __restrict__ XT,
                                                int mb, int tid) {
#pragma unroll
  for (int i = 0; i < 4; ++i) {
    int c = tid + 128 * i;          // 0..511 chunks of 8 halves
    int f = c >> 1, half = (c & 1) * 8;
    uint32_t laddr = (uint32_t)(uintptr_t)(dst_lds + f * 32 + half);
    uint64_t gaddr = (uint64_t)(uintptr_t)(XT + (size_t)f * NN + mb + half);
    asm volatile("global_load_async_to_lds_b128 %0, %1, off"
                 :: "v"(laddr), "v"(gaddr)
                 : "memory");
  }
}

// ---- fused normalize + relation write (g==NG-1) + agg = P·X ---------------

__global__ __launch_bounds__(128) void agg_kernel(
    const _Float16* __restrict__ th, const _Float16* __restrict__ ph,
    const _Float16* __restrict__ XT, const float* __restrict__ pos,
    const float* __restrict__ rowinv, float* __restrict__ rel_out,
    _Float16* __restrict__ aggh) {
  // double-buffered XT chunk: [buf][f (256)][m' (32)] f16  (2 x 16 KB)
  __shared__ __attribute__((aligned(64))) _Float16 xbuf[2][NFG * 32];
  __shared__ __attribute__((aligned(32))) _Float16 plds[4][16 * 32];
  int tid  = threadIdx.x;
  int wave = tid >> 5, lane = tid & 31;
  int lo = lane & 15, hi = lane >> 4;
  int g  = blockIdx.y;
  int n0 = blockIdx.x * 64 + wave * 16;
  const _Float16* thg = th + (size_t)g * NN * NFR;
  const _Float16* phg = ph + (size_t)g * NN * NFR;
  _Float16* myl = &plds[wave][0];
  const bool writeRel = (g == NG - 1);

  v16h a0 = load_a16(thg, NFR, n0, 0,  lane);
  v16h a1 = load_a16(thg, NFR, n0, 32, lane);

  float px[8], py[8], rin[8];
#pragma unroll
  for (int j = 0; j < 8; ++j) {
    int n = n0 + j + 8 * hi;
    px[j] = pos[2 * n]; py[j] = pos[2 * n + 1];
    rin[j] = rowinv[g * NN + n];
  }

  v8f acc[16] = {};   // agg rows: 16 f-tiles of 16 columns

  // prologue: stage first chunk
  fill_xbuf_async(&xbuf[0][0], XT, 0, tid);

  for (int mb = 0; mb < NN; mb += 32) {
    int cur = (mb >> 5) & 1;
    // own async copy into xbuf[cur] complete; barrier makes it block-visible
    // and guarantees everyone is done reading xbuf[cur^1].
    asm volatile("s_wait_asynccnt 0x0" ::: "memory");
    __syncthreads();
    if (mb + 32 < NN)
      fill_xbuf_async(&xbuf[cur ^ 1][0], XT, mb + 32, tid);

#pragma unroll
    for (int sub = 0; sub < 2; ++sub) {
      int m0 = mb + sub * 16;
      v16h b0 = load_b16(phg, NFR, m0, 0,  lane);
      v16h b1 = load_b16(phg, NFR, m0, 32, lane);
      v8f c = {};
      c = wmma_f16(a0, b0, c);
      c = wmma_f16(a1, b1, c);
      float qx = pos[2 * (m0 + lo)], qy = pos[2 * (m0 + lo) + 1];
#pragma unroll
      for (int j = 0; j < 8; ++j) {
        float dx = px[j] - qx, dy = py[j] - qy;
        float p = (dx * dx + dy * dy > RADIUS2)
                      ? 0.0f
                      : __expf(c[j] * SCALE) * rin[j];
        if (writeRel)
          rel_out[(size_t)(n0 + j + 8 * hi) * NN + m0 + lo] = p;
        myl[(j + 8 * hi) * 32 + sub * 16 + lo] = (_Float16)p;  // P staging
      }
    }
    // wave-local DS ordering: stores above complete before loads below
    asm volatile("s_wait_dscnt 0x0" ::: "memory");
    // read P tile back as A operand (16x32, ld=32)
    v16h pa;
    {
      const _Float16* p = myl + lo * 32 + 16 * hi;
      v8h x = *(const v8h*)(p);
      v8h y = *(const v8h*)(p + 8);
      pa = __builtin_shufflevector(x, y,
                                   0,1,2,3,4,5,6,7,8,9,10,11,12,13,14,15);
    }
    const _Float16* xb = &xbuf[cur][0];
#pragma unroll
    for (int ft = 0; ft < 16; ++ft) {
      // B(k=m', f) = XT chunk [f][m'] staged in LDS
      v16h b = *(const v16h*)(xb + (ft * 16 + lo) * 32 + 16 * hi);
      acc[ft] = wmma_f16(pa, b, acc[ft]);
    }
  }

#pragma unroll
  for (int ft = 0; ft < 16; ++ft)
#pragma unroll
    for (int j = 0; j < 8; ++j)
      aggh[((size_t)g * NN + n0 + j + 8 * hi) * NFG + ft * 16 + lo] =
          (_Float16)acc[ft][j];
}

// ---- proj = agg·gcn_wᵀ, LayerNorm, ReLU, sum over g -----------------------

__global__ __launch_bounds__(64) void proj_kernel(
    const _Float16* __restrict__ aggh, const _Float16* __restrict__ gcnh,
    const float* __restrict__ lng, const float* __restrict__ lnb,
    float* __restrict__ out) {
  __shared__ float outl[32 * 256];   // 32 KB, per-wave regions
  int wave = threadIdx.x >> 5, lane = threadIdx.x & 31;
  int lo = lane & 15, hi = lane >> 4;
  int n0 = blockIdx.x * 32 + wave * 16;
  float* myout = &outl[wave * 16 * 256];

  for (int i = lane; i < 16 * 256; i += 32) myout[i] = 0.0f;

  for (int g = 0; g < NG; ++g) {
    const _Float16* A = aggh + (size_t)g * NN * NFG;
    const _Float16* W = gcnh + (size_t)g * NFG * NFG;  // [o][f] row-major
    v8f acc[16] = {};
    for (int kk = 0; kk < NFG; kk += 32) {
      v16h a = load_a16(A, NFG, n0, kk, lane);
#pragma unroll
      for (int ot = 0; ot < 16; ++ot) {
        v16h b = load_b16(W, NFG, ot * 16, kk, lane);  // B(f,o)=W[o][f]
        acc[ot] = wmma_f16(a, b, acc[ot]);
      }
    }
    // LayerNorm statistics per row
    float mean[8], rstd[8];
#pragma unroll
    for (int j = 0; j < 8; ++j) {
      float s = 0.0f;
#pragma unroll
      for (int ot = 0; ot < 16; ++ot) s += acc[ot][j];
      for (int off = 1; off < 16; off <<= 1) s += __shfl_xor(s, off, 32);
      mean[j] = s * (1.0f / NFG);
      float q = 0.0f;
#pragma unroll
      for (int ot = 0; ot < 16; ++ot) {
        float d = acc[ot][j] - mean[j];
        q += d * d;
      }
      for (int off = 1; off < 16; off <<= 1) q += __shfl_xor(q, off, 32);
      rstd[j] = rsqrtf(q * (1.0f / NFG) + LN_EPS);
    }
    // scale/shift + ReLU, accumulate into LDS
#pragma unroll
    for (int ot = 0; ot < 16; ++ot) {
      int o = ot * 16 + lo;
      float gv = lng[g * NFG + o], bv = lnb[g * NFG + o];
#pragma unroll
      for (int j = 0; j < 8; ++j) {
        float v = (acc[ot][j] - mean[j]) * rstd[j] * gv + bv;
        v = v > 0.0f ? v : 0.0f;
        myout[(j + 8 * hi) * 256 + o] += v;
      }
    }
  }
  asm volatile("s_wait_dscnt 0x0" ::: "memory");
  for (int i = lane; i < 16 * 256; i += 32)
    out[(size_t)n0 * 256 + i] = myout[i];
}

// ---------------------------------------------------------------------------

extern "C" void kernel_launch(void* const* d_in, const int* in_sizes, int n_in,
                              void* d_out, int out_size, void* d_ws,
                              size_t ws_size, hipStream_t stream) {
  (void)in_sizes; (void)n_in; (void)out_size; (void)ws_size;
  const float* X       = (const float*)d_in[0];
  const float* boxes   = (const float*)d_in[1];
  const float* theta_w = (const float*)d_in[2];
  const float* theta_b = (const float*)d_in[3];
  const float* phi_w   = (const float*)d_in[4];
  const float* phi_b   = (const float*)d_in[5];
  const float* gcn_w   = (const float*)d_in[6];
  const float* ln_g    = (const float*)d_in[7];
  const float* ln_b    = (const float*)d_in[8];

  float* out = (float*)d_out;                        // [N, NFG]
  float* rel = out + (size_t)NN * NFG;               // [N, N] relation[-1]

  char* w = (char*)d_ws;
  auto alloc = [&](size_t bytes) -> char* {
    char* p = w;
    w += (bytes + 255) & ~(size_t)255;
    return p;
  };
  float*    pos    = (float*)   alloc((size_t)NN * 2 * 4);
  _Float16* Xh     = (_Float16*)alloc((size_t)NN * NFG * 2);
  _Float16* XT     = (_Float16*)alloc((size_t)NFG * NN * 2);
  _Float16* thetah = (_Float16*)alloc((size_t)NG * NFR * NFG * 2);
  _Float16* phih   = (_Float16*)alloc((size_t)NG * NFR * NFG * 2);
  _Float16* gcnh   = (_Float16*)alloc((size_t)NG * NFG * NFG * 2);
  _Float16* thh    = (_Float16*)alloc((size_t)NG * NN * NFR * 2);
  _Float16* phh    = (_Float16*)alloc((size_t)NG * NN * NFR * 2);
  float*    rowinv = (float*)   alloc((size_t)NG * NN * 4);
  _Float16* aggh   = (_Float16*)alloc((size_t)NG * NN * NFG * 2);

  pos_kernel<<<NN / 256, 256, 0, stream>>>(boxes, pos);
  cvtXT_kernel<<<(NN * NFG) / 256, 256, 0, stream>>>(X, Xh, XT);
  cvt_kernel<<<(NG * NFR * NFG) / 256, 256, 0, stream>>>(theta_w, thetah,
                                                         NG * NFR * NFG);
  cvt_kernel<<<(NG * NFR * NFG) / 256, 256, 0, stream>>>(phi_w, phih,
                                                         NG * NFR * NFG);
  cvt_kernel<<<(NG * NFG * NFG) / 256, 256, 0, stream>>>(gcn_w, gcnh,
                                                         NG * NFG * NFG);

  dim3 gridG(NN / 64, NG);
  thph_kernel<<<gridG, 128, 0, stream>>>(Xh, thetah, theta_b, thh);
  thph_kernel<<<gridG, 128, 0, stream>>>(Xh, phih, phi_b, phh);
  denom_kernel<<<gridG, 128, 0, stream>>>(thh, phh, pos, rowinv);
  agg_kernel<<<gridG, 128, 0, stream>>>(thh, phh, XT, pos, rowinv, rel, aggh);
  proj_kernel<<<NN / 32, 64, 0, stream>>>(aggh, gcnh, ln_g, ln_b, out);
}
